// VGRNN_50328426774823
// MI455X (gfx1250) — matrix operand order repaired and checked by reading, not compiled, not run on hardware
//
#include <hip/hip_runtime.h>
#include <hip/hip_bf16.h>

typedef __attribute__((ext_vector_type(16))) _Float16 v16h;
typedef __attribute__((ext_vector_type(8)))  float    v8f;

// ---------------------------------------------------------------------------
// float atomic-max via sign-split int/uint atomics (works for any sign mix)
// ---------------------------------------------------------------------------
__device__ __forceinline__ void atomicMaxFloat(float* addr, float val) {
    if (val >= 0.0f)
        atomicMax((int*)addr, __float_as_int(val));
    else
        atomicMin((unsigned int*)addr, __float_as_uint(val));
}

// ---------------------------------------------------------------------------
// q/k/v/skip linear: one wave = 16 node rows, 4 linears x 2 col-tiles
// = 8 x v_wmma_f32_16x16x32_f16 per wave. W is [32 out][32 in] row-major,
// out = in @ W^T + b.
// ---------------------------------------------------------------------------
__global__ __launch_bounds__(256) void lin_qkvs_kernel(
    const float* __restrict__ in,
    const float* __restrict__ Wq, const float* __restrict__ bq,
    const float* __restrict__ Wk, const float* __restrict__ bk,
    const float* __restrict__ Wv, const float* __restrict__ bv,
    const float* __restrict__ Ws, const float* __restrict__ bs,
    float* __restrict__ oq, float* __restrict__ ok,
    float* __restrict__ ov, float* __restrict__ os, int n)
{
    const int wave = blockIdx.x * (blockDim.x >> 5) + (threadIdx.x >> 5);
    const int lane = threadIdx.x & 31;
    const int mrow = lane & 15;
    const int hi   = lane >> 4;          // 0: lanes 0-15, 1: lanes 16-31
    const int row0 = wave * 16;
    if (row0 >= n) return;               // wave-uniform exit: EXEC stays all-1s
    const bool full = (row0 + 16 <= n);  // wave-uniform (always true for N%16==0)

    // ---- A fragment: 16x32 f16, lane holds row (row0+mrow) ----
    // K slots: a[0..7] = K=(hi?8:0)+i ; a[8..15] = K=(hi?24:16)+i
    int arow = row0 + mrow; if (arow >= n) arow = n - 1;
    const float* ap = in + (size_t)arow * 32 + hi * 8;
    const float4 a0 = *(const float4*)(ap + 0);
    const float4 a1 = *(const float4*)(ap + 4);
    const float4 a2 = *(const float4*)(ap + 16);
    const float4 a3 = *(const float4*)(ap + 20);
    v16h a;
    a[0]  = (_Float16)a0.x; a[1]  = (_Float16)a0.y; a[2]  = (_Float16)a0.z; a[3]  = (_Float16)a0.w;
    a[4]  = (_Float16)a1.x; a[5]  = (_Float16)a1.y; a[6]  = (_Float16)a1.z; a[7]  = (_Float16)a1.w;
    a[8]  = (_Float16)a2.x; a[9]  = (_Float16)a2.y; a[10] = (_Float16)a2.z; a[11] = (_Float16)a2.w;
    a[12] = (_Float16)a3.x; a[13] = (_Float16)a3.y; a[14] = (_Float16)a3.z; a[15] = (_Float16)a3.w;

    const float* Wt[4] = {Wq, Wk, Wv, Ws};
    const float* Bt[4] = {bq, bk, bv, bs};
    float*       Ot[4] = {oq, ok, ov, os};

#pragma unroll
    for (int t = 0; t < 4; ++t) {
#pragma unroll
        for (int nt = 0; nt < 2; ++nt) {
            const int ncol = nt * 16 + mrow;                 // output channel
            // ---- B fragment: B[k][n] = W[n][k]; lane holds col ncol,
            //      K slots: b[0..7] = K=(hi?8:0)+i ; b[8..15] = K=(hi?24:16)+i ----
            const float* wp = Wt[t] + (size_t)ncol * 32 + hi * 8;
            const float4 w0 = *(const float4*)(wp + 0);
            const float4 w1 = *(const float4*)(wp + 4);
            const float4 w2 = *(const float4*)(wp + 16);
            const float4 w3 = *(const float4*)(wp + 20);
            v16h bf;
            bf[0]  = (_Float16)w0.x; bf[1]  = (_Float16)w0.y; bf[2]  = (_Float16)w0.z; bf[3]  = (_Float16)w0.w;
            bf[4]  = (_Float16)w1.x; bf[5]  = (_Float16)w1.y; bf[6]  = (_Float16)w1.z; bf[7]  = (_Float16)w1.w;
            bf[8]  = (_Float16)w2.x; bf[9]  = (_Float16)w2.y; bf[10] = (_Float16)w2.z; bf[11] = (_Float16)w2.w;
            bf[12] = (_Float16)w3.x; bf[13] = (_Float16)w3.y; bf[14] = (_Float16)w3.z; bf[15] = (_Float16)w3.w;

            const float bias = Bt[t][ncol];
            v8f c;
#pragma unroll
            for (int i = 0; i < 8; ++i) c[i] = bias;         // bias depends only on N

            c = __builtin_amdgcn_wmma_f32_16x16x32_f16(
                    false, a, false, bf, (short)0, c, false, false);

            // D layout: VGPR v -> row (row0 + 8*hi + v), col ncol
            float* op = Ot[t] + (size_t)(row0 + hi * 8) * 32 + ncol;
            if (full) {
#pragma unroll
                for (int v = 0; v < 8; ++v) op[(size_t)v * 32] = c[v];  // imm-offset stores
            } else {
#pragma unroll
                for (int v = 0; v < 8; ++v) {
                    int r = row0 + hi * 8 + v;
                    if (r < n) op[(size_t)v * 32] = c[v];
                }
            }
        }
    }
}

// ---------------------------------------------------------------------------
// per-conv state init: accum=0, m=-inf, ssum=0
// ---------------------------------------------------------------------------
__global__ void conv_init_kernel(float* __restrict__ accum, float* __restrict__ mbuf,
                                 float* __restrict__ ssum, int n)
{
    int i = blockIdx.x * blockDim.x + threadIdx.x;
    if (i < n) { mbuf[i] = -__builtin_inff(); ssum[i] = 0.0f; }
    if (i < n * 32) accum[i] = 0.0f;
}

// ---------------------------------------------------------------------------
// edge pass 1: logits + per-destination max. One wave per edge, lane=channel.
// ---------------------------------------------------------------------------
__global__ __launch_bounds__(256) void edge_logits_kernel(
    const int* __restrict__ src, const int* __restrict__ dst,
    const float* __restrict__ q, const float* __restrict__ k,
    float* __restrict__ logits, float* __restrict__ mbuf, int ne)
{
    const int lane = threadIdx.x & 31;
    int wid = blockIdx.x * (blockDim.x >> 5) + (threadIdx.x >> 5);
    int nw  = gridDim.x * (blockDim.x >> 5);
    for (int e = wid; e < ne; e += nw) {
        int s = src[e], d = dst[e];
        float p = q[(size_t)d * 32 + lane] * k[(size_t)s * 32 + lane];
#pragma unroll
        for (int off = 16; off; off >>= 1) p += __shfl_xor(p, off, 32);
        if (lane == 0) {
            float lg = p * 0.17677669529663687f;   // 1/sqrt(32)
            logits[e] = lg;
            atomicMaxFloat(mbuf + d, lg);
        }
    }
}

// ---------------------------------------------------------------------------
// edge pass 2: w = exp(l - m[dst]); accumulate sum-of-w and unnormalized
// sum of w * v[src] (coalesced 128B atomic burst per edge).
// ---------------------------------------------------------------------------
__global__ __launch_bounds__(256) void edge_agg_kernel(
    const int* __restrict__ src, const int* __restrict__ dst,
    const float* __restrict__ vmat, const float* __restrict__ logits,
    const float* __restrict__ mbuf, float* __restrict__ ssum,
    float* __restrict__ accum, int ne)
{
    const int lane = threadIdx.x & 31;
    int wid = blockIdx.x * (blockDim.x >> 5) + (threadIdx.x >> 5);
    int nw  = gridDim.x * (blockDim.x >> 5);
    for (int e = wid; e < ne; e += nw) {
        int s = src[e], d = dst[e];
        float w = __expf(logits[e] - mbuf[d]);
        if (lane == 0) atomicAdd(ssum + d, w);
        atomicAdd(accum + (size_t)d * 32 + lane, w * vmat[(size_t)s * 32 + lane]);
    }
}

// ---------------------------------------------------------------------------
// conv_out = accum / (ssum + 1e-16) + skip ; write or accumulate into preact
// ---------------------------------------------------------------------------
__global__ void finalize_kernel(const float* __restrict__ accum, const float* __restrict__ ssum,
                                const float* __restrict__ skip, float* preact, int n, int mode)
{
    int i = blockIdx.x * blockDim.x + threadIdx.x;
    if (i >= n * 32) return;
    int node = i >> 5;
    float val = accum[i] / (ssum[node] + 1e-16f) + skip[i];
    preact[i] = mode ? (preact[i] + val) : val;
}

// ---------------------------------------------------------------------------
// gates: z = sigmoid(zpre) (in place), rh = sigmoid(rpre) * h
// ---------------------------------------------------------------------------
__global__ void gates_kernel(float* zpre, const float* __restrict__ rpre,
                             const float* __restrict__ h, float* __restrict__ rh, int n)
{
    int i = blockIdx.x * blockDim.x + threadIdx.x;
    if (i >= n * 32) return;
    float z = 1.0f / (1.0f + __expf(-zpre[i]));
    float r = 1.0f / (1.0f + __expf(-rpre[i]));
    zpre[i] = z;
    rh[i] = r * h[i];
}

// ---------------------------------------------------------------------------
// out = z*h + (1-z)*tanh(htpre)
// ---------------------------------------------------------------------------
__global__ void output_kernel(const float* __restrict__ zg, const float* __restrict__ h,
                              const float* __restrict__ htpre, float* __restrict__ out, int n)
{
    int i = blockIdx.x * blockDim.x + threadIdx.x;
    if (i >= n * 32) return;
    float z = zg[i];
    out[i] = z * h[i] + (1.0f - z) * tanhf(htpre[i]);
}

// ---------------------------------------------------------------------------
extern "C" void kernel_launch(void* const* d_in, const int* in_sizes, int n_in,
                              void* d_out, int out_size, void* d_ws, size_t ws_size,
                              hipStream_t stream)
{
    const int N = in_sizes[0] / 32;
    const int E = in_sizes[2] / 2;

    const float* x  = (const float*)d_in[0];
    const float* h  = (const float*)d_in[1];
    const int*   ei = (const int*)d_in[2];
    const int* esrc = ei;
    const int* edst = ei + E;

    // side 0 = x weights (d_in[3..10]), side 1 = h weights (d_in[11..18]);
    // order per side: (Wq,bq),(Wk,bk),(Wv,bv),(Ws,bs); each W [3,32,32], b [3,32]
    const float* W[2][4]; const float* B[2][4];
    for (int s = 0; s < 2; ++s)
        for (int t = 0; t < 4; ++t) {
            W[s][t] = (const float*)d_in[3 + s * 8 + t * 2];
            B[s][t] = (const float*)d_in[3 + s * 8 + t * 2 + 1];
        }

    // workspace layout (floats): ~61 MB total
    float* ws  = (float*)d_ws;
    size_t nc  = (size_t)N * 32;
    float* q      = ws;
    float* kk     = q + nc;
    float* vv     = kk + nc;
    float* sk     = vv + nc;
    float* accum  = sk + nc;
    float* mbuf   = accum + nc;
    float* ssum   = mbuf + N;
    float* logits = ssum + N;
    float* zpre   = logits + E;    // becomes z gate in place
    float* rpre   = zpre + nc;
    float* htpre  = rpre + nc;
    float* rh     = htpre + nc;

    dim3 blk(256);
    int linBlocks  = ((N + 15) / 16 + 7) / 8;       // 8 waves/block, 16 rows/wave
    int nodeBlocks = (int)((nc + 255) / 256);
    int edgeBlocks = (E + 7) / 8;                   // 8 waves/block, 1 edge/wave

    auto conv = [&](const float* inp, int side, int ci, float* target, int mode) {
        lin_qkvs_kernel<<<linBlocks, blk, 0, stream>>>(inp,
            W[side][0] + ci * 1024, B[side][0] + ci * 32,
            W[side][1] + ci * 1024, B[side][1] + ci * 32,
            W[side][2] + ci * 1024, B[side][2] + ci * 32,
            W[side][3] + ci * 1024, B[side][3] + ci * 32,
            q, kk, vv, sk, N);
        conv_init_kernel<<<nodeBlocks, blk, 0, stream>>>(accum, mbuf, ssum, N);
        edge_logits_kernel<<<edgeBlocks, blk, 0, stream>>>(esrc, edst, q, kk, logits, mbuf, E);
        edge_agg_kernel<<<edgeBlocks, blk, 0, stream>>>(esrc, edst, vv, logits, mbuf, ssum, accum, E);
        finalize_kernel<<<nodeBlocks, blk, 0, stream>>>(accum, ssum, sk, target, N, mode);
    };

    conv(x, 0, 0, zpre, 0);   // cx(0, x)
    conv(h, 1, 0, zpre, 1);   // ch(0, h)          -> z_pre
    conv(x, 0, 1, rpre, 0);   // cx(1, x)
    conv(h, 1, 1, rpre, 1);   // ch(1, h)          -> r_pre
    conv(x, 0, 2, htpre, 0);  // cx(2, x)

    gates_kernel<<<nodeBlocks, blk, 0, stream>>>(zpre, rpre, h, rh, N);

    conv(rh, 1, 2, htpre, 1); // ch(2, r_g * h)    -> h_tilde pre-activation

    output_kernel<<<nodeBlocks, blk, 0, stream>>>(zpre, h, htpre, (float*)d_out, N);
}
